// PatchEncoder_52166672777654
// MI455X (gfx1250) — matrix operand order repaired and checked by reading, not compiled, not run on hardware
//
#include <hip/hip_runtime.h>

typedef _Float16 v16h __attribute__((ext_vector_type(16)));
typedef float    v8f  __attribute__((ext_vector_type(8)));
typedef float    f4   __attribute__((ext_vector_type(4)));

#define Hn 512
#define Wn 512

// One block = one coarse 16x16 output patch (16 rows x 16 cols of pixels).
// 8 waves x 2 segments; each segment = ONE v_wmma_f32_16x16x32_f16 computing
// conv3x3 (27 taps padded to K=32) for 16 pixels x 3 channels.
// All fragment-build memory ops use compile-time immediate offsets.
__global__ __launch_bounds__(256)
void patch_encoder_conv_wmma(const float* __restrict__ X,
                             const float* __restrict__ Wk,   // [3][3][3][3] HWIO
                             const float* __restrict__ bias, // [3]
                             const float* __restrict__ pos,  // [16384][48]
                             float* __restrict__ out)        // [32][1024][768]
{
    __shared__ float tile[18 * 18 * 3];   // haloed f32 input tile (972)
    __shared__ f4    ob4[192];            // staged results (768 floats)
    float* ob = (float*)ob4;

    const int tid  = threadIdx.x;
    const int wv   = tid >> 5;            // wave 0..7
    const int lane = tid & 31;

    int bid = blockIdx.x;
    const int wb = bid & 31;  bid >>= 5;  // 32 patch cols
    const int hb = bid & 31;  bid >>= 5;  // 32 patch rows
    const int b  = bid;                   // batch
    const int h0 = hb << 4;
    const int w0 = wb << 4;

    // ---- cooperative haloed tile load: 324 pixels (rows/cols h0-1..h0+16)
    for (int t = tid; t < 324; t += 256) {
        int r  = t / 18;
        int cl = t - r * 18;
        int gh = h0 - 1 + r;
        int gw = w0 - 1 + cl;
        float v0 = 0.0f, v1 = 0.0f, v2 = 0.0f;   // SAME conv zero pad
        if (gh >= 0 && gh < Hn && gw >= 0 && gw < Wn) {
            const float* p = X + ((long)(b * Hn + gh) * Wn + gw) * 3;
            v0 = p[0]; v1 = p[1]; v2 = p[2];     // merges to global_load_b96
        }
        tile[t * 3 + 0] = v0;
        tile[t * 3 + 1] = v1;
        tile[t * 3 + 2] = v2;
    }
    __syncthreads();

    const int m  = lane & 15;    // A/C row (pixel within segment)
    const int hi = lane >> 4;    // lane half selects which K values we hold
    const int n  = m;            // B/C column (output channel; only n<3 live)

    // ---- bias folded into C (channel-constant across all rows)
    const float bv = (n < 3) ? bias[(n < 3) ? n : 2] : 0.0f;

    // ---- B fragment (shared by both segments): 32x16 f16
    // lanes 0-15: K=0..15 ; lanes 16-31: K=16..31 for column N=n
    v16h bm;
    #pragma unroll
    for (int e = 0; e < 16; ++e) bm[e] = (_Float16)0.0f;
    if (n < 3) {
        if (hi == 0) {
            #pragma unroll
            for (int e = 0; e < 16; ++e)          // K = e (compile-time)
                bm[e] = (_Float16)Wk[e * 3 + n];
        } else {
            #pragma unroll
            for (int e = 0; e < 11; ++e)          // K = 16+e, valid K<27
                bm[e] = (_Float16)Wk[(16 + e) * 3 + n];
        }
    }

    // ---- two row-segments per wave: rows 2*wv and 2*wv+1
    #pragma unroll
    for (int s = 0; s < 2; ++s) {
        const int sr   = (wv << 1) + s;           // row within patch, 0..15
        const int base = sr * 54 + m * 3;         // f32 index of window start

        // A fragment: 16-bit A 16x32 layout. tap t=(ky*3+kx)*3+ci maps to
        // tile offset (t/9)*54 + t%9 from base — compile-time per element.
        v16h a;
        if (hi == 0) {
            // K=e -> t=e
            #pragma unroll
            for (int e = 0; e < 16; ++e) {
                int t = e;
                a[e] = (_Float16)tile[base + (t / 9) * 54 + (t % 9)];
            }
        } else {
            // K=e+16 -> t=e+16 (zero for t>=27)
            #pragma unroll
            for (int e = 0; e < 16; ++e) {
                int t = e + 16;
                a[e] = (t < 27) ? (_Float16)tile[base + (t / 9) * 54 + (t % 9)]
                                : (_Float16)0.0f;
            }
        }

        v8f c;
        #pragma unroll
        for (int r = 0; r < 8; ++r) c[r] = bv;

        c = __builtin_amdgcn_wmma_f32_16x16x32_f16(false, a, false, bm,
                                                   (short)0, c, false, false);

        // stage (C layout: lane l, VGPR r -> M=r+8*(l>>4), N=l&15)
        if (n < 3) {
            const int ib = sr * 48 + (hi << 4) + (hi << 3) + n; // +24*hi
            #pragma unroll
            for (int r = 0; r < 8; ++r)
                ob[ib + r * 3] = c[r];
        }
    }
    __syncthreads();

    // ---- store: one contiguous 768-float patch, pos_emb added as float4.
    // j = ph*48 + pw*3 + c ; pos k = (h&3)*12 + (pw&3)*3 + c advances in
    // lockstep within a fine 4x4 patch; 4-aligned j-groups never cross the
    // 12-float fine-patch boundary, so both sides are contiguous float4.
    if (tid < 192) {
        const int j0  = tid << 2;
        const int ph  = j0 / 48;
        const int r1  = j0 - ph * 48;
        const int qc  = r1 / 12;            // fine-patch column within patch
        const int sub = r1 - qc * 12;       // offset within fine patch row
        const int h   = h0 + ph;
        const int nf  = (h >> 2) * 128 + (w0 >> 2) + qc;
        const f4* pos4 = (const f4*)pos;
        f4 pe = pos4[nf * 12 + (h & 3) * 3 + (sub >> 2)];
        f4 o  = *(const f4*)&ob4[tid];
        o[0] += pe[0]; o[1] += pe[1]; o[2] += pe[2]; o[3] += pe[3];

        const int np = (hb << 5) + wb;
        f4* out4 = (f4*)(out + (long)(b * 1024 + np) * 768);
        __builtin_nontemporal_store(o, &out4[tid]);  // write-once: keep L2 for X
    }
}

extern "C" void kernel_launch(void* const* d_in, const int* in_sizes, int n_in,
                              void* d_out, int out_size, void* d_ws, size_t ws_size,
                              hipStream_t stream) {
    (void)in_sizes; (void)n_in; (void)d_ws; (void)ws_size; (void)out_size;
    const float* X    = (const float*)d_in[0];
    const float* Wk   = (const float*)d_in[1];
    const float* bias = (const float*)d_in[2];
    const float* pos  = (const float*)d_in[3];
    float* out = (float*)d_out;

    dim3 grid(32 * 32 * 32);   // B * (H/16) * (W/16) = 32768 blocks (1/patch)
    dim3 block(256);           // 8 waves
    hipLaunchKernelGGL(patch_encoder_conv_wmma, grid, block, 0, stream,
                       X, Wk, bias, pos, out);
}
